// GMMNet_U_12945031430857
// MI455X (gfx1250) — matrix operand order repaired and checked by reading, not compiled, not run on hardware
//
#include <hip/hip_runtime.h>
#include <hip/hip_bf16.h>
#include <math.h>

// ---------------------------------------------------------------------------
// GMMNet-U (4x GMMConv, 64ch, K=3) for MI455X / gfx1250 (wave32, WMMA).
//
// Strategy:
//   * Hoist per-edge GEMM to per-node:  y[N,256] = x[N,64] @ [g(64x192) | root(64x64)]
//     via v_wmma_f32_16x16x32_f16 (f16 in, f32 accumulate, K=64 -> 2 WMMAs/tile).
//   * Edge phase: one wave per edge. Gaussian kernel weights w_k computed from
//     edge_attr / mu / sigma on the fly (exact expf). Gather 192 f32 of xg[src]
//     (38MB table -> resident in 192MB L2), scatter 64 global_atomic_add_f32
//     into agg[dst] (12.8MB -> L2-resident atomics).
//   * Fused epilogue: mean-div, +xr, +bias, optional skip (U-net), exact-erf
//     GELU, re-zero agg for next layer, emit f16 activations for next GEMM.
// ---------------------------------------------------------------------------

typedef __attribute__((ext_vector_type(16))) _Float16 v16h;
typedef __attribute__((ext_vector_type(8)))  _Float16 v8h;
typedef __attribute__((ext_vector_type(8)))  float    v8f;

#define GN_N 50000
#define GN_E 800000
#define GN_CIN 64
#define GN_W 256          // 192 (g) + 64 (root)
#define GN_EPS 1e-15f

// ---------------------------------------------------------------------------
// init: zero agg + cnt, convert x (f32) -> xh (f16)
__global__ void k_init(const float* __restrict__ x, _Float16* __restrict__ xh,
                       float* __restrict__ agg, float* __restrict__ cnt) {
  int i = blockIdx.x * blockDim.x + threadIdx.x;
  if (i < GN_N * GN_CIN) {
    agg[i] = 0.0f;
    xh[i] = (_Float16)x[i];
    if (i < GN_N) cnt[i] = 0.0f;
  }
}

// in-degree histogram (dst is edge_index row 1)
__global__ void k_count(const int* __restrict__ ei, float* __restrict__ cnt) {
  int e = blockIdx.x * blockDim.x + threadIdx.x;
  if (e < GN_E) atomicAdd(&cnt[ei[GN_E + e]], 1.0f);
}

// ---------------------------------------------------------------------------
// Pack [g | root] (f32, row-major [K=64, 256]) into this code's wave32 WMMA B-operand
// register layout: pw[tileN(16)][kb(2)][lane(32)][j(16)] halfs.
//   lane l holds column n = tileN*16 + (l&15), K = kb*32 + (l>>4)*16 + j.
__global__ void k_pack(const float* __restrict__ g, const float* __restrict__ root,
                       _Float16* __restrict__ pw) {
  int i = blockIdx.x * blockDim.x + threadIdx.x;     // 16*2*32*16 = 16384
  if (i >= 16 * 2 * 32 * 16) return;
  int j  = i & 15;
  int l  = (i >> 4) & 31;
  int kb = (i >> 9) & 1;
  int t  = i >> 10;
  int n  = t * 16 + (l & 15);
  int k  = kb * 32 + ((l >> 4) << 4) + j;
  float v = (n < 192) ? g[k * 192 + n] : root[k * 64 + (n - 192)];
  pw[i] = (_Float16)v;
}

// ---------------------------------------------------------------------------
// y[N,256] = xh[N,64] @ pw   (one 16x16 output tile per wave, K=64 -> 2 WMMAs)
// grid = (N/16, 2), block = 256 (8 waves); wave w handles n-tile blockIdx.y*8+w.
__global__ void k_gemm(const _Float16* __restrict__ xh,
                       const _Float16* __restrict__ pw,
                       float* __restrict__ y) {
  const int lane = threadIdx.x & 31;
  const int wid  = threadIdx.x >> 5;
  const int tm   = blockIdx.x;                 // 0..3124
  const int tn   = blockIdx.y * 8 + wid;       // 0..15
  const int row  = tm * 16 + (lane & 15);      // A row this lane supplies
  const int khalf = (lane >> 4) * 16;          // lanes 0-15: K 0-15; 16-31: 16-31

  v8f c = {};
  const v16h* pB = (const v16h*)pw + (size_t)(tn * 2) * 32 + lane;
#pragma unroll
  for (int kb = 0; kb < 2; ++kb) {
    const _Float16* pa = xh + (size_t)row * GN_CIN + kb * 32 + khalf;
    v8h lo = *(const v8h*)(pa);
    v8h hi = *(const v8h*)(pa + 8);
    v16h a = __builtin_shufflevector(lo, hi, 0, 1, 2, 3, 4, 5, 6, 7,
                                             8, 9, 10, 11, 12, 13, 14, 15);
    v16h b = pB[kb * 32];
    c = __builtin_amdgcn_wmma_f32_16x16x32_f16(
        /*neg_a=*/false, a, /*neg_b=*/false, b,
        /*c_mod=*/(short)0, c, /*reuse_a=*/false, /*reuse_b=*/false);
  }
  // C/D layout: VGPR r, lane l -> row (r + 8*(l>>4)), col (l&15)
  const int col = tn * 16 + (lane & 15);
  const int r0  = tm * 16 + (lane >> 4) * 8;
#pragma unroll
  for (int r = 0; r < 8; ++r) y[(size_t)(r0 + r) * GN_W + col] = c[r];
}

// ---------------------------------------------------------------------------
// Edge phase: one wave per edge, lane = channel pair {lane, lane+32}.
__global__ void k_edge(const int* __restrict__ ei, const float* __restrict__ ea,
                       const float* __restrict__ mu, const float* __restrict__ sigma,
                       const float* __restrict__ y, float* __restrict__ agg) {
  const int lane = threadIdx.x & 31;
  const int wid  = threadIdx.x >> 5;
  const int e    = blockIdx.x * (blockDim.x >> 5) + wid;
  if (e >= GN_E) return;                       // wave-uniform

  const int src = ei[e];
  const int dst = ei[GN_E + e];
  const float a = ea[e];

  float w[3];
#pragma unroll
  for (int k = 0; k < 3; ++k) {
    float d = a - mu[k];
    float s = sigma[k];
    w[k] = expf(-0.5f * d * d / (GN_EPS + s * s));
  }

  const float* yr = y + (size_t)src * GN_W;    // xg part = cols 0..191 (L2-resident)
  float m0 = w[0] * yr[lane]      + w[1] * yr[64 + lane]  + w[2] * yr[128 + lane];
  float m1 = w[0] * yr[32 + lane] + w[1] * yr[96 + lane]  + w[2] * yr[160 + lane];

  float* ar = agg + (size_t)dst * GN_CIN;      // L2 atomics, no return needed
  atomicAdd(ar + lane,      m0);
  atomicAdd(ar + 32 + lane, m1);
}

// ---------------------------------------------------------------------------
// Epilogue: mean, + x@root, + bias, optional skip, optional exact GELU.
// Also re-zeroes agg (next layer / next call) and emits f16 activations.
__global__ void k_epi(float* __restrict__ agg, const float* __restrict__ y,
                      const float* __restrict__ cnt, const float* __restrict__ bias,
                      const float* __restrict__ skip, float* __restrict__ dstF,
                      _Float16* __restrict__ xh_out, int doGelu) {
  int i = blockIdx.x * blockDim.x + threadIdx.x;
  if (i >= GN_N * GN_CIN) return;
  int n = i >> 6, c = i & 63;
  float v = agg[i] / fmaxf(cnt[n], 1.0f) + y[(size_t)n * GN_W + 192 + c] + bias[c];
  if (skip) v += skip[i];
  if (doGelu) v = 0.5f * v * (1.0f + erff(v * 0.70710678118654752440f));
  if (dstF) dstF[i] = v;
  if (xh_out) xh_out[i] = (_Float16)v;
  agg[i] = 0.0f;
}

// ---------------------------------------------------------------------------
extern "C" void kernel_launch(void* const* d_in, const int* in_sizes, int n_in,
                              void* d_out, int out_size, void* d_ws, size_t ws_size,
                              hipStream_t stream) {
  (void)in_sizes; (void)n_in; (void)out_size; (void)ws_size;

  const float* x   = (const float*)d_in[0];
  const int*   ei  = (const int*)d_in[1];
  const float* ea  = (const float*)d_in[2];
  // params: 4 layers x (g, mu, sigma, root, bias) starting at d_in[3]
  const float* g_[4], *mu_[4], *sg_[4], *rt_[4], *bs_[4];
  for (int l = 0; l < 4; ++l) {
    g_[l]  = (const float*)d_in[3 + 5 * l + 0];
    mu_[l] = (const float*)d_in[3 + 5 * l + 1];
    sg_[l] = (const float*)d_in[3 + 5 * l + 2];
    rt_[l] = (const float*)d_in[3 + 5 * l + 3];
    bs_[l] = (const float*)d_in[3 + 5 * l + 4];
  }

  // workspace layout (256B-aligned slices)
  auto AL = [](size_t s) { return (s + 255) & ~(size_t)255; };
  char* ws = (char*)d_ws;
  size_t off = 0;
  _Float16* xh  = (_Float16*)(ws + off); off += AL((size_t)GN_N * GN_CIN * 2);
  float*    y   = (float*)   (ws + off); off += AL((size_t)GN_N * GN_W   * 4);
  float*    agg = (float*)   (ws + off); off += AL((size_t)GN_N * GN_CIN * 4);
  float*    cnt = (float*)   (ws + off); off += AL((size_t)GN_N * 4);
  float*    h0  = (float*)   (ws + off); off += AL((size_t)GN_N * GN_CIN * 4);
  _Float16* pw  = (_Float16*)(ws + off); off += AL((size_t)4 * 16384 * 2);

  const int NC = GN_N * GN_CIN;                     // 3.2M
  k_init<<<dim3((NC + 255) / 256), dim3(256), 0, stream>>>(x, xh, agg, cnt);
  k_count<<<dim3((GN_E + 255) / 256), dim3(256), 0, stream>>>(ei, cnt);
  for (int l = 0; l < 4; ++l)
    k_pack<<<dim3(64), dim3(256), 0, stream>>>(g_[l], rt_[l], pw + (size_t)l * 16384);

  // U-net schedule: L0 gelu->h0, L1 gelu, L2 (+h0) gelu, L3 plain -> d_out
  float*    dstF[4]  = { h0, nullptr, nullptr, (float*)d_out };
  const float* skp[4] = { nullptr, nullptr, h0, nullptr };
  _Float16* xho[4]   = { xh, xh, xh, nullptr };
  int       gel[4]   = { 1, 1, 1, 0 };

  const dim3 gGemm(GN_N / 16, 2), bGemm(256);       // 8 waves/block, 16 n-tiles
  const dim3 gEdge(GN_E / 8), bEdge(256);           // 8 edges/block (wave each)
  const dim3 gEpi((NC + 255) / 256), bEpi(256);

  for (int l = 0; l < 4; ++l) {
    k_gemm<<<gGemm, bGemm, 0, stream>>>(xh, pw + (size_t)l * 16384, y);
    k_edge<<<gEdge, bEdge, 0, stream>>>(ei, ea, mu_[l], sg_[l], y, agg);
    k_epi <<<gEpi,  bEpi,  0, stream>>>(agg, y, cnt, bs_[l], skp[l], dstF[l],
                                        xho[l], gel[l]);
  }
}
